// GP_Layer_28578712388008
// MI455X (gfx1250) — matrix operand order repaired and checked by reading, not compiled, not run on hardware
//
#include <hip/hip_runtime.h>
#include <hip/hip_bf16.h>
#include <stdint.h>

// ---------------------------------------------------------------------------
// GP layer: x_l = lin(x,w1,b1); x_r = lin(x,w2,b2); gp = Cayley(x_l,x_r);
//           y = lin(gp,w3,b3); out = mv_layernorm(y, a_norm)
// N=4096, C=512, 8 blades. Blade-major bf16 planes; WMMA f32_16x16x32_bf16
// GEMMs with ping-pong LDS and (if available) async global->LDS copies.
// ---------------------------------------------------------------------------

typedef __attribute__((ext_vector_type(16))) __bf16 v16bf;
typedef __attribute__((ext_vector_type(8)))  float  v8f;
typedef unsigned int u32x4 __attribute__((ext_vector_type(4)));
typedef int v4i __attribute__((ext_vector_type(4)));

#define N_ROWS 4096
#define CH     512
#define PLANE  (N_ROWS * CH)   // elements per blade plane
#define WPLANE (CH * CH)       // elements per sub-weight plane

#define BM 64
#define BN 128
#define BK 32
#define ROWB 40                // ushorts per LDS row: 32 data + 8 pad (80B)
#define NTHR 128               // 4 waves

__device__ __constant__ int d_sub[8] = {0, 1, 1, 1, 2, 2, 2, 3};

__device__ inline unsigned short f2bf(float f) {
  unsigned u = __float_as_uint(f);
  u += 0x7FFFu + ((u >> 16) & 1u);       // round to nearest even
  return (unsigned short)(u >> 16);
}
__device__ inline float bf2f(unsigned short h) {
  return __uint_as_float(((unsigned)h) << 16);
}

union FragU { u32x4 q[2]; v16bf v; };

// ----- CDNA5 async global->LDS copy (ASYNCcnt) with compile-time fallback ---
// Builtin signature (from hipcc diagnostic): (int4 __device__*, int4 __shared__*,
// imm offset, imm cpol). Use explicit target-AS typed pointer casts.
#if __has_builtin(__builtin_amdgcn_global_load_async_to_lds_b128)
#define ASYNC_CP 1
typedef __attribute__((address_space(1))) v4i* gv4i_p;
typedef __attribute__((address_space(3))) v4i* lv4i_p;
__device__ inline void async_cp16(const void* g, void* l) {
  __builtin_amdgcn_global_load_async_to_lds_b128((gv4i_p)g, (lv4i_p)l, 0, 0);
}
__device__ inline void async_wait0() {
#if __has_builtin(__builtin_amdgcn_s_wait_asynccnt)
  __builtin_amdgcn_s_wait_asynccnt(0);
#else
  asm volatile("s_wait_asynccnt 0x0" ::: "memory");
#endif
}
#else
#define ASYNC_CP 0
#endif

// ---------------------------------------------------------------------------
// Pack x (N,512,8) f32 -> blade-major bf16 planes [8][N][512]
// ---------------------------------------------------------------------------
__global__ __launch_bounds__(256)
void conv_x(const float* __restrict__ x, unsigned short* __restrict__ Xb) {
  const int p = blockIdx.x * 256 + threadIdx.x;   // over PLANE
  const float4 lo = *(const float4*)(x + (size_t)p * 8);
  const float4 hi = *(const float4*)(x + (size_t)p * 8 + 4);
  const float v[8] = {lo.x, lo.y, lo.z, lo.w, hi.x, hi.y, hi.z, hi.w};
#pragma unroll
  for (int i = 0; i < 8; ++i)
    Xb[(size_t)i * PLANE + p] = f2bf(v[i]);
}

// Pack w (512,512,4) f32 -> sub-major bf16 planes [4][512(n)][512(m)]
__global__ __launch_bounds__(256)
void conv_w(const float* __restrict__ w, unsigned short* __restrict__ Wb) {
  const int p = blockIdx.x * 256 + threadIdx.x;   // over WPLANE (n*512+m)
  const float4 s = *(const float4*)(w + (size_t)p * 4);
  Wb[p]              = f2bf(s.x);
  Wb[WPLANE + p]     = f2bf(s.y);
  Wb[2 * WPLANE + p] = f2bf(s.z);
  Wb[3 * WPLANE + p] = f2bf(s.w);
}

// ---------------------------------------------------------------------------
// Blade-GEMM: Out[side][blade] = A[blade](NxK) * W[side][sub(blade)]^T(KxN')
// grid = (N/BM, 512/BN, nz), z = side*8 + blade (side via z>>3).
// 128 threads (4 waves); wave owns 16x128 strip = 8 accumulator tiles.
// Ping-pong LDS; next k-tile fetched (async if available) under current math.
// ---------------------------------------------------------------------------
template <bool OUT_BF16>
__global__ __launch_bounds__(NTHR)
void gemm_blade(const unsigned short* __restrict__ Aall,
                const unsigned short* __restrict__ W1p,
                const unsigned short* __restrict__ W2p,
                const float* __restrict__ bias1,
                const float* __restrict__ bias2,
                void* __restrict__ Out1,
                void* __restrict__ Out2) {
  __shared__ __align__(16) unsigned short sA[2][BM * ROWB];
  __shared__ __align__(16) unsigned short sB[2][BN * ROWB];

  const int z     = blockIdx.z;
  const int blade = z & 7;
  const int side  = z >> 3;
  const int sub   = d_sub[blade];
  const unsigned short* Ap = Aall + (size_t)blade * PLANE;                 // [N][512]
  const unsigned short* Wp = (side ? W2p : W1p) + (size_t)sub * WPLANE;    // [512][512]
  const float* bias = side ? bias2 : bias1;
  void* OutAll      = side ? Out2 : Out1;

  const int tid  = threadIdx.x;
  const int wave = tid >> 5;
  const int lane = tid & 31;
  const int m16  = lane & 15;
  const int half = lane >> 4;

  const int gm0 = blockIdx.x * BM;
  const int gn0 = blockIdx.y * BN;

  v8f acc[8];
#pragma unroll
  for (int c = 0; c < 8; ++c)
#pragma unroll
    for (int e = 0; e < 8; ++e) acc[c][e] = 0.f;

  // Loaders: A = 2 threads/row (32B each); B = 1 thread/row (64B)
  const int arowl = tid >> 1;
  const int apart = tid & 1;
  const unsigned short* gA = Ap + (size_t)(gm0 + arowl) * CH + apart * 16;
  const unsigned short* gB = Wp + (size_t)(gn0 + tid) * CH;
  const int lAoff = arowl * ROWB + apart * 16;
  const int lBoff = tid * ROWB;

#if ASYNC_CP
  // ---- prologue: tile 0 via async copy ----
  async_cp16(gA,      &sA[0][lAoff]);
  async_cp16(gA + 8,  &sA[0][lAoff + 8]);
  async_cp16(gB,      &sB[0][lBoff]);
  async_cp16(gB + 8,  &sB[0][lBoff + 8]);
  async_cp16(gB + 16, &sB[0][lBoff + 16]);
  async_cp16(gB + 24, &sB[0][lBoff + 24]);
  gA += BK; gB += BK;
  async_wait0();
  __syncthreads();
#else
  // ---- prologue: tile 0 via register staging ----
  {
    u32x4 ra0 = *(const u32x4*)(gA);
    u32x4 ra1 = *(const u32x4*)(gA + 8);
    u32x4 rb0 = *(const u32x4*)(gB);
    u32x4 rb1 = *(const u32x4*)(gB + 8);
    u32x4 rb2 = *(const u32x4*)(gB + 16);
    u32x4 rb3 = *(const u32x4*)(gB + 24);
    gA += BK; gB += BK;
    *(u32x4*)(&sA[0][lAoff])      = ra0;
    *(u32x4*)(&sA[0][lAoff + 8])  = ra1;
    *(u32x4*)(&sB[0][lBoff])      = rb0;
    *(u32x4*)(&sB[0][lBoff + 8])  = rb1;
    *(u32x4*)(&sB[0][lBoff + 16]) = rb2;
    *(u32x4*)(&sB[0][lBoff + 24]) = rb3;
  }
  __syncthreads();
#endif

  int cur = 0;
  for (int k0 = 0; k0 < CH; k0 += BK) {
    const bool hasNext = (k0 + BK) < CH;
    const int  nxt     = cur ^ 1;

#if ASYNC_CP
    if (hasNext) {   // fetch tile k+1 straight into the other LDS buffer
      async_cp16(gA,      &sA[nxt][lAoff]);
      async_cp16(gA + 8,  &sA[nxt][lAoff + 8]);
      async_cp16(gB,      &sB[nxt][lBoff]);
      async_cp16(gB + 8,  &sB[nxt][lBoff + 8]);
      async_cp16(gB + 16, &sB[nxt][lBoff + 16]);
      async_cp16(gB + 24, &sB[nxt][lBoff + 24]);
      gA += BK; gB += BK;
    }
#else
    u32x4 ra0, ra1, rb0, rb1, rb2, rb3;
    if (hasNext) {   // fetch tile k+1 into registers
      ra0 = *(const u32x4*)(gA);
      ra1 = *(const u32x4*)(gA + 8);
      rb0 = *(const u32x4*)(gB);
      rb1 = *(const u32x4*)(gB + 8);
      rb2 = *(const u32x4*)(gB + 16);
      rb3 = *(const u32x4*)(gB + 24);
      gA += BK; gB += BK;
      __builtin_prefetch(gA, 0, 3);
      __builtin_prefetch(gB, 0, 3);
    }
#endif

    // Load A fragment + ALL 8 B fragments first, then the WMMA burst,
    // so ds latency pipelines instead of serializing per tile.
    FragU af;
    const unsigned short* arow = &sA[cur][(wave * 16 + m16) * ROWB];
    af.q[0] = *(const u32x4*)(arow + half * 8);
    af.q[1] = *(const u32x4*)(arow + 16 + half * 8);
    FragU bf[8];
#pragma unroll
    for (int c = 0; c < 8; ++c) {
      const unsigned short* brow = &sB[cur][(c * 16 + m16) * ROWB + half * 8];
      bf[c].q[0] = *(const u32x4*)(brow);
      bf[c].q[1] = *(const u32x4*)(brow + 16);
    }
#pragma unroll
    for (int c = 0; c < 8; ++c)
      acc[c] = __builtin_amdgcn_wmma_f32_16x16x32_bf16(
          false, af.v, false, bf[c].v, (short)0, acc[c], false, false);

    if (hasNext) {
#if ASYNC_CP
      async_wait0();            // next tile landed in LDS
#else
      *(u32x4*)(&sA[nxt][lAoff])      = ra0;
      *(u32x4*)(&sA[nxt][lAoff + 8])  = ra1;
      *(u32x4*)(&sB[nxt][lBoff])      = rb0;
      *(u32x4*)(&sB[nxt][lBoff + 8])  = rb1;
      *(u32x4*)(&sB[nxt][lBoff + 16]) = rb2;
      *(u32x4*)(&sB[nxt][lBoff + 24]) = rb3;
#endif
      __syncthreads();
      cur = nxt;
    }
  }

  // Epilogue: D layout — lane (m16, half): row = half*8 + v, col = m16
#pragma unroll
  for (int c = 0; c < 8; ++c) {
    const int gn = gn0 + c * 16 + m16;
    const float bv = (blade == 0) ? bias[gn] : 0.f;
#pragma unroll
    for (int v = 0; v < 8; ++v) {
      const int gm = gm0 + wave * 16 + half * 8 + v;
      const float val = acc[c][v] + bv;
      if (OUT_BF16)
        ((unsigned short*)OutAll)[(size_t)blade * PLANE + (size_t)gm * CH + gn] = f2bf(val);
      else
        ((float*)OutAll)[(size_t)blade * PLANE + (size_t)gm * CH + gn] = val;
    }
  }
}

// ---------------------------------------------------------------------------
// Pointwise geometric product via Cayley table (64 signed terms), 2 elems/thread
// ---------------------------------------------------------------------------
__global__ __launch_bounds__(256)
void geo_product(const unsigned short* __restrict__ XL,
                 const unsigned short* __restrict__ XR,
                 unsigned short* __restrict__ GP) {
  const int p = blockIdx.x * 256 + threadIdx.x;   // over PLANE/2 element-pairs
  const int idx[8] = {0, 1, 2, 4, 3, 5, 6, 7};    // blade value -> ordered index

  float l0[8], l1[8], r0[8], r1[8];
#pragma unroll
  for (int i = 0; i < 8; ++i) {
    const unsigned lv = *(const unsigned*)(XL + (size_t)i * PLANE + 2 * (size_t)p);
    const unsigned rv = *(const unsigned*)(XR + (size_t)i * PLANE + 2 * (size_t)p);
    l0[i] = bf2f((unsigned short)lv);  l1[i] = bf2f((unsigned short)(lv >> 16));
    r0[i] = bf2f((unsigned short)rv);  r1[i] = bf2f((unsigned short)(rv >> 16));
  }

  float g0[8], g1[8];
#pragma unroll
  for (int j = 0; j < 8; ++j) { g0[j] = 0.f; g1[j] = 0.f; }

#pragma unroll
  for (int a = 0; a < 8; ++a) {
#pragma unroll
    for (int b = 0; b < 8; ++b) {
      int swaps = 0;
      for (int sh = 1; (a >> sh); ++sh) swaps += __popc((a >> sh) & b);
      const float s  = (swaps & 1) ? -1.f : 1.f;
      const int   j  = idx[a ^ b];
      g0[j] += s * l0[idx[a]] * r0[idx[b]];
      g1[j] += s * l1[idx[a]] * r1[idx[b]];
    }
  }

#pragma unroll
  for (int j = 0; j < 8; ++j) {
    const unsigned o = (unsigned)f2bf(g0[j]) | ((unsigned)f2bf(g1[j]) << 16);
    *(unsigned*)(GP + (size_t)j * PLANE + 2 * (size_t)p) = o;
  }
}

// ---------------------------------------------------------------------------
// Multivector layernorm; metric (ordered basis) = {1,1,1,1,-1,-1,-1,-1}
// ---------------------------------------------------------------------------
__global__ __launch_bounds__(256)
void mv_ln(const float* __restrict__ Y, const float* __restrict__ a_norm,
           float* __restrict__ out) {
  const int r   = blockIdx.x;
  const int tid = threadIdx.x;
  const float metric[8] = {1.f, 1.f, 1.f, 1.f, -1.f, -1.f, -1.f, -1.f};

  float y[2][8];
  float partial = 0.f;
#pragma unroll
  for (int s = 0; s < 2; ++s) {
    const int c = tid + s * 256;
    float q = 0.f;
#pragma unroll
    for (int i = 0; i < 8; ++i) {
      const float v = Y[(size_t)i * PLANE + (size_t)r * CH + c];
      y[s][i] = v;
      q += metric[i] * v * v;
    }
    partial += sqrtf(sqrtf(q * q + 1e-16f));
  }

  __shared__ float red[256];
  red[tid] = partial;
  __syncthreads();
#pragma unroll
  for (int off = 128; off > 0; off >>= 1) {
    if (tid < off) red[tid] += red[tid + off];
    __syncthreads();
  }
  const float inv = 1.f / (red[0] * (1.f / 512.f) + 1e-6f);

#pragma unroll
  for (int s = 0; s < 2; ++s) {
    const int c = tid + s * 256;
    const float sc = a_norm[c] * inv;
    float* dst = out + ((size_t)r * CH + c) * 8;
    const float4 o0 = make_float4(sc * y[s][0], sc * y[s][1], sc * y[s][2], sc * y[s][3]);
    const float4 o1 = make_float4(sc * y[s][4], sc * y[s][5], sc * y[s][6], sc * y[s][7]);
    *(float4*)(dst)     = o0;
    *(float4*)(dst + 4) = o1;
  }
}

// ---------------------------------------------------------------------------
extern "C" void kernel_launch(void* const* d_in, const int* in_sizes, int n_in,
                              void* d_out, int out_size, void* d_ws, size_t ws_size,
                              hipStream_t stream) {
  const float* x  = (const float*)d_in[0];
  const float* w1 = (const float*)d_in[1];
  const float* b1 = (const float*)d_in[2];
  const float* w2 = (const float*)d_in[3];
  const float* b2 = (const float*)d_in[4];
  const float* w3 = (const float*)d_in[5];
  const float* b3 = (const float*)d_in[6];
  const float* an = (const float*)d_in[7];

  // Workspace: Xb 32MB | Wb 6MB | XLb 32MB | XRb 32MB | GPb 32MB
  // Y (f32, 64MB) reuses XLb+XRb after geo_product consumed them.
  char* ws = (char*)d_ws;
  unsigned short* Xb  = (unsigned short*)(ws);
  unsigned short* Wb  = (unsigned short*)(ws + (size_t)8 * PLANE * 2);
  unsigned short* XLb = (unsigned short*)(ws + (size_t)8 * PLANE * 2 + (size_t)12 * WPLANE * 2);
  unsigned short* XRb = XLb + (size_t)8 * PLANE;
  unsigned short* GPb = XRb + (size_t)8 * PLANE;
  float* Y = (float*)XLb;

  conv_x<<<PLANE / 256, 256, 0, stream>>>(x, Xb);
  conv_w<<<WPLANE / 256, 256, 0, stream>>>(w1, Wb);
  conv_w<<<WPLANE / 256, 256, 0, stream>>>(w2, Wb + 4 * WPLANE);
  conv_w<<<WPLANE / 256, 256, 0, stream>>>(w3, Wb + 8 * WPLANE);

  // x_l and x_r in a single launch: z = side*8 + blade
  gemm_blade<true ><<<dim3(N_ROWS / BM, CH / BN, 16), NTHR, 0, stream>>>(
      Xb, Wb, Wb + 4 * WPLANE, b1, b2, (void*)XLb, (void*)XRb);
  geo_product<<<(PLANE / 2) / 256, 256, 0, stream>>>(XLb, XRb, GPb);
  gemm_blade<false><<<dim3(N_ROWS / BM, CH / BN, 8), NTHR, 0, stream>>>(
      GPb, Wb + 8 * WPLANE, Wb + 8 * WPLANE, b3, b3, (void*)Y, (void*)Y);
  mv_ln<<<N_ROWS, 256, 0, stream>>>(Y, an, (float*)d_out);
}
